// LongformerAttentionWrapperWithRotary_85134841741609
// MI455X (gfx1250) — matrix operand order repaired and checked by reading, not compiled, not run on hardware
//
#include <hip/hip_runtime.h>
#include <hip/hip_bf16.h>
#include <stdint.h>

#define H_     16
#define D_     128
#define ROT_   32
#define WIN_   256
#define G_     64
#define SEQ_   2048
#define B_     2
#define HD_    (H_ * D_)       // 2048
#define QKVLD  (3 * HD_)       // 6144
#define SQ_    (SEQ_ - WIN_)   // 1792
#define PADS   2080            // SEQ_ padded so the 288-key band never runs OOB
#define SCALE_ 0.08838834764831845f
#define MINV_  (-3.40282347e38f)

typedef __attribute__((ext_vector_type(16))) __bf16         v16bf;
typedef __attribute__((ext_vector_type(8)))  float          v8f;
typedef __attribute__((ext_vector_type(8)))  unsigned short u16x8;

union FragU {
    struct { u16x8 lo, hi; } s;
    v16bf v;
};

__device__ __forceinline__ unsigned short f32_to_bf16(float f) {
    unsigned u = __float_as_uint(f);
    return (unsigned short)((u + 0x7FFFu + ((u >> 16) & 1u)) >> 16);   // RNE
}

// CDNA5 async global->LDS copy (16B per lane, tracked by ASYNCcnt).
// INST_OFFSET is added to BOTH the LDS and global addresses (ISA 08 §4.4).
#define GLOBAL_ASYNC_TO_LDS_B128(lds_u32, gaddr_u64, OFFSTR)               \
    asm volatile("global_load_async_to_lds_b128 %0, %1, off" OFFSTR       \
                 :: "v"(lds_u32), "v"(gaddr_u64) : "memory")
#define WAIT_ASYNCCNT0() asm volatile("s_wait_asynccnt 0x0" ::: "memory")

// ---------------------------------------------------------------- fp32 -> bf16 (plain)
__global__ __launch_bounds__(256)
void k_f32_to_bf16(const float* __restrict__ x, unsigned short* __restrict__ y,
                   long long n) {
    long long i      = (long long)blockIdx.x * blockDim.x + threadIdx.x;
    long long stride = (long long)gridDim.x * blockDim.x;
    for (; i < n; i += stride) y[i] = f32_to_bf16(x[i]);
}

// ---------------------------------------------------------------- fp32 -> bf16 + transpose
__global__ __launch_bounds__(256)
void k_f32_to_bf16_T(const float* __restrict__ W, unsigned short* __restrict__ Wt,
                     int K, int N) {
    __shared__ unsigned short tile[32][33];
    int n0 = blockIdx.x * 32;
    int k0 = blockIdx.y * 32;
    int tx = threadIdx.x;   // 0..31
    int ty = threadIdx.y;   // 0..7
#pragma unroll
    for (int j = 0; j < 4; ++j) {
        int k = k0 + ty + j * 8;
        tile[ty + j * 8][tx] = f32_to_bf16(W[(size_t)k * N + n0 + tx]);
    }
    __syncthreads();
#pragma unroll
    for (int j = 0; j < 4; ++j) {
        int n = n0 + ty + j * 8;
        Wt[(size_t)n * K + k0 + tx] = tile[tx][ty + j * 8];
    }
}

// ---------------------------------------------------------------- WMMA GEMM
// C[M,N] = A[M,K](bf16) * Bt[N,K](bf16) + bias[N].  Double-buffered LDS with
// ASYNC global->LDS staging; 8 waves in 4(m) x 2(n); wave tile 32x64 = 8 WMMA.
#define BM  128
#define BN  128
#define BKk 32
#define LDT (BKk + 8)

__global__ __launch_bounds__(256)
void k_gemm_bf16(const unsigned short* __restrict__ A,
                 const unsigned short* __restrict__ Bt,
                 const float* __restrict__ bias,
                 float* __restrict__ C, int M, int N, int K) {
    __shared__ unsigned short Asl[2][BM][LDT];
    __shared__ unsigned short Bsl[2][BN][LDT];

    const int tid  = threadIdx.x;
    const int wv   = tid >> 5;
    const int lane = tid & 31;
    const int hl   = lane >> 4;
    const int r16  = lane & 15;
    const int m0   = blockIdx.y * BM;
    const int n0   = blockIdx.x * BN;
    const int wm   = (wv & 3) * 32;
    const int wn   = (wv >> 2) * 64;

    v8f acc[2][4];
#pragma unroll
    for (int mi = 0; mi < 2; ++mi)
#pragma unroll
        for (int nt = 0; nt < 4; ++nt)
#pragma unroll
            for (int j = 0; j < 8; ++j) acc[mi][nt][j] = 0.0f;

    const int row = tid >> 1;
    const int seg = (tid & 1) * 16;
    const unsigned short* agBase = A  + (size_t)(m0 + row) * K + seg;
    const unsigned short* bgBase = Bt + (size_t)(n0 + row) * K + seg;
    const unsigned ldsA0 = (unsigned)(size_t)&Asl[0][row][seg];
    const unsigned ldsA1 = (unsigned)(size_t)&Asl[1][row][seg];
    const unsigned ldsB0 = (unsigned)(size_t)&Bsl[0][row][seg];
    const unsigned ldsB1 = (unsigned)(size_t)&Bsl[1][row][seg];

    // ---- prologue: async-stage tile 0 into buffer 0
    {
        unsigned long long ga = (unsigned long long)agBase;
        unsigned long long gb = (unsigned long long)bgBase;
        GLOBAL_ASYNC_TO_LDS_B128(ldsA0, ga, "");
        GLOBAL_ASYNC_TO_LDS_B128(ldsA0, ga, " offset:16");
        GLOBAL_ASYNC_TO_LDS_B128(ldsB0, gb, "");
        GLOBAL_ASYNC_TO_LDS_B128(ldsB0, gb, " offset:16");
    }
    WAIT_ASYNCCNT0();
    __syncthreads();

    int buf = 0;
    for (int k0 = 0; k0 < K; k0 += BKk) {
        const bool hn = (k0 + BKk) < K;
        if (hn) {   // async-stage next tile into alternate buffer; hides behind WMMAs
            unsigned long long ga = (unsigned long long)(agBase + k0 + BKk);
            unsigned long long gb = (unsigned long long)(bgBase + k0 + BKk);
            unsigned la = buf ? ldsA0 : ldsA1;
            unsigned lb = buf ? ldsB0 : ldsB1;
            GLOBAL_ASYNC_TO_LDS_B128(la, ga, "");
            GLOBAL_ASYNC_TO_LDS_B128(la, ga, " offset:16");
            GLOBAL_ASYNC_TO_LDS_B128(lb, gb, "");
            GLOBAL_ASYNC_TO_LDS_B128(lb, gb, " offset:16");
            if (k0 + 2 * BKk < K) {
                __builtin_prefetch(agBase + k0 + 2 * BKk, 0, 1);
                __builtin_prefetch(bgBase + k0 + 2 * BKk, 0, 1);
            }
        }

        FragU ua[2], ub[4];
#pragma unroll
        for (int mi = 0; mi < 2; ++mi) {
            ua[mi].s.lo = *(const u16x8*)&Asl[buf][wm + mi * 16 + r16][hl * 8];
            ua[mi].s.hi = *(const u16x8*)&Asl[buf][wm + mi * 16 + r16][16 + hl * 8];
        }
#pragma unroll
        for (int nt = 0; nt < 4; ++nt) {
            ub[nt].s.lo = *(const u16x8*)&Bsl[buf][wn + nt * 16 + r16][hl * 8];
            ub[nt].s.hi = *(const u16x8*)&Bsl[buf][wn + nt * 16 + r16][16 + hl * 8];
        }
#pragma unroll
        for (int mi = 0; mi < 2; ++mi)
#pragma unroll
            for (int nt = 0; nt < 4; ++nt)
                acc[mi][nt] = __builtin_amdgcn_wmma_f32_16x16x32_bf16(
                    false, ua[mi].v, false, ub[nt].v, (short)0, acc[mi][nt],
                    false, false);

        WAIT_ASYNCCNT0();   // this wave's async LDS writes landed
        __syncthreads();    // …and are visible block-wide
        buf ^= 1;
    }

#pragma unroll
    for (int mi = 0; mi < 2; ++mi)
#pragma unroll
        for (int nt = 0; nt < 4; ++nt) {
            int gn   = n0 + wn + nt * 16 + r16;
            float bv = bias[gn];
#pragma unroll
            for (int i = 0; i < 8; ++i) {
                int gm = m0 + wm + mi * 16 + hl * 8 + i;
                C[(size_t)gm * N + gn] = acc[mi][nt][i] + bv;
            }
        }
}

// ---------------------------------------------------------------- RoPE + head mean
__global__ __launch_bounds__(128)
void k_rope_mean(float* __restrict__ qkv, float* __restrict__ kmean,
                 float* __restrict__ vmean) {
    int blk = blockIdx.x;
    int b = blk / SEQ_, s = blk % SEQ_;
    int d = threadIdx.x;

    float c = 1.0f, sn = 0.0f;
    if (d < ROT_) {
        float invf = __powf(10000.0f, -(float)(d & 15) / 16.0f);
        float fr   = (float)s * invf;
        c  = __cosf(fr);
        sn = __sinf(fr);
    }

    float* base = qkv + (size_t)(b * SEQ_ + s) * QKVLD;
    float ksum = 0.0f, vsum = 0.0f;
    for (int h = 0; h < H_; ++h) {
        float* p  = base + h * (3 * D_);
        float kv  = p[D_ + d];
        float vv  = p[2 * D_ + d];
        float qv = 0.f, qp = 0.f, kp = 0.f;
        if (d < ROT_) {
            qv = p[d];
            qp = p[d ^ 16];
            kp = p[D_ + (d ^ 16)];
        }
        __syncthreads();
        if (d < ROT_) {
            float rq = (d < 16) ? -qp : qp;
            float rk = (d < 16) ? -kp : kp;
            p[d]      = qv * c + rq * sn;
            kv        = kv * c + rk * sn;
            p[D_ + d] = kv;
        }
        ksum += kv;
        vsum += vv;
        __syncthreads();
    }
    size_t mo = (size_t)(b * SEQ_ + s) * D_ + d;
    kmean[mo] = ksum * (1.0f / H_);
    vmean[mo] = vsum * (1.0f / H_);
}

// ---------------------------------------------------------------- bf16 attn operand prep
__global__ __launch_bounds__(128)
void k_prep_kv(const float* __restrict__ kmean, const float* __restrict__ vmean,
               unsigned short* __restrict__ kbf, unsigned short* __restrict__ vbfT) {
    int blk = blockIdx.x;
    int b = blk / PADS, s = blk % PADS;
    int d = threadIdx.x;
    unsigned short kv = 0, vv = 0;
    if (s < SEQ_) {
        size_t src = (size_t)(b * SEQ_ + s) * D_ + d;
        kv = f32_to_bf16(kmean[src]);
        vv = f32_to_bf16(vmean[src]);
    }
    kbf[(size_t)(b * PADS + s) * D_ + d] = kv;
    vbfT[((size_t)b * D_ + d) * PADS + s] = vv;
}

__global__ __launch_bounds__(128)
void k_prep_q(const float* __restrict__ qkv, unsigned short* __restrict__ qbf) {
    int idx = blockIdx.x;
    int s  = idx % SEQ_;
    int bh = idx / SEQ_;
    int h = bh % H_, b = bh / H_;
    int d = threadIdx.x;
    qbf[((size_t)(b * H_ + h) * SEQ_ + s) * D_ + d] =
        f32_to_bf16(qkv[(size_t)(b * SEQ_ + s) * QKVLD + h * (3 * D_) + d]);
}

// ---------------------------------------------------------------- helpers (VALU prefix)
__device__ __forceinline__ float dot128(const float4* __restrict__ q4,
                                        const float* __restrict__ krow) {
    const float4* k4 = (const float4*)krow;
    float acc = 0.0f;
#pragma unroll
    for (int i = 0; i < 32; ++i) {
        float4 a = q4[i], bb = k4[i];
        acc += a.x * bb.x + a.y * bb.y + a.z * bb.z + a.w * bb.w;
    }
    return acc;
}

// ---------------------------------------------------------------- prefix attention (VALU)
__global__ __launch_bounds__(256)
void k_attn_prefix(const float* __restrict__ qkv, const float* __restrict__ kmean,
                   const float* __restrict__ vmean, const float* __restrict__ amask,
                   float* __restrict__ out) {
    __shared__ float qv[8][128];
    __shared__ float sc[8][WIN_];

    int tid = threadIdx.x, w = tid >> 5, lane = tid & 31;
    int qblk = blockIdx.x % (WIN_ / 8);
    int bh   = blockIdx.x / (WIN_ / 8);
    int h = bh % H_, b = bh / H_;
    int qpos = qblk * 8 + w;

    const float* qp = qkv + (size_t)(b * SEQ_ + qpos) * QKVLD + h * (3 * D_);
    for (int d = lane; d < 128; d += 32) qv[w][d] = qp[d];
    __syncthreads();

    const float* kb = kmean + (size_t)b * SEQ_ * D_;
    const float* vb = vmean + (size_t)b * SEQ_ * D_;
    const float* am = amask + (size_t)b * SEQ_;
    const float4* q4 = (const float4*)&qv[w][0];

    for (int key = lane; key < WIN_; key += 32) {
        float s = (key <= qpos) ? dot128(q4, kb + (size_t)key * D_) * SCALE_ : MINV_;
        sc[w][key] = s + am[key];
    }
    __syncthreads();

    float m = MINV_;
    for (int key = lane; key < WIN_; key += 32) m = fmaxf(m, sc[w][key]);
#pragma unroll
    for (int off = 16; off >= 1; off >>= 1) m = fmaxf(m, __shfl_xor(m, off));
    float sum = 0.0f;
    for (int key = lane; key < WIN_; key += 32) {
        float e = __expf(sc[w][key] - m);
        sc[w][key] = e;
        sum += e;
    }
#pragma unroll
    for (int off = 16; off >= 1; off >>= 1) sum += __shfl_xor(sum, off);
    __syncthreads();

    float inv = 1.0f / sum;
    float acc0 = 0.f, acc1 = 0.f, acc2 = 0.f, acc3 = 0.f;
    for (int key = 0; key < WIN_; ++key) {
        float wk = sc[w][key];
        const float* vr = vb + (size_t)key * D_;
        acc0 += wk * vr[lane];
        acc1 += wk * vr[lane + 32];
        acc2 += wk * vr[lane + 64];
        acc3 += wk * vr[lane + 96];
    }
    float* op = out + (size_t)(b * SEQ_ + qpos) * HD_ + h * D_;
    op[lane]      = acc0 * inv;
    op[lane + 32] = acc1 * inv;
    op[lane + 64] = acc2 * inv;
    op[lane + 96] = acc3 * inv;
}

// ---------------------------------------------------------------- step attention (WMMA flash)
__global__ __launch_bounds__(256)
void k_attn_steps_wmma(const unsigned short* __restrict__ qbf,
                       const unsigned short* __restrict__ kbf,
                       const unsigned short* __restrict__ vbfT,
                       const float* __restrict__ amask,
                       const int* __restrict__ gidx,
                       float* __restrict__ out) {
    __shared__ unsigned short VgT[128][72];    // compacted V_glob^T [dim][g]
    __shared__ unsigned short Pl[8][16][40];   // per-wave P scratch (C->A layout)

    const int tid  = threadIdx.x;
    const int w    = tid >> 5;
    const int lane = tid & 31;
    const int hl   = lane >> 4;
    const int r16  = lane & 15;

    const int tile  = blockIdx.x;
    const int hg    = blockIdx.y;
    const int b     = blockIdx.z;
    const int h     = hg * 8 + w;
    const int tbase = tile * 16;

    const int*   gb = gidx + b * G_;
    const float* am = amask + (size_t)b * SEQ_;
    const unsigned short* kb = kbf  + (size_t)b * PADS * D_;
    const unsigned short* vt = vbfT + (size_t)b * D_ * PADS;

    for (int idx = tid; idx < 128 * G_; idx += 256) {
        int d = idx >> 6;
        int g = idx & 63;
        VgT[d][g] = vt[(size_t)d * PADS + gb[g]];
    }
    __syncthreads();

    const unsigned short* qrow =
        qbf + (((size_t)(b * H_ + h) * SEQ_) + (WIN_ + tbase)) * D_;
    FragU uq[4];
#pragma unroll
    for (int kq = 0; kq < 4; ++kq) {
        uq[kq].s.lo = *(const u16x8*)(qrow + (size_t)r16 * D_ + kq * 32 + hl * 8);
        uq[kq].s.hi = *(const u16x8*)(qrow + (size_t)r16 * D_ + kq * 32 + 16 + hl * 8);
    }

    float mrow[8], lrow[8];
    v8f oacc[8];
#pragma unroll
    for (int i = 0; i < 8; ++i) { mrow[i] = MINV_; lrow[i] = 0.0f; }
#pragma unroll
    for (int nt = 0; nt < 8; ++nt)
#pragma unroll
        for (int i = 0; i < 8; ++i) oacc[nt][i] = 0.0f;

    for (int c = 0; c < 11; ++c) {
        const bool isg = (c < 2);
        int gv[2];
        if (isg) {
            gv[0] = gb[c * 32 + r16];
            gv[1] = gb[c * 32 + 16 + r16];
        }

        v8f st[2];
#pragma unroll
        for (int t = 0; t < 2; ++t) {
#pragma unroll
            for (int i = 0; i < 8; ++i) st[t][i] = 0.0f;
            int krow = isg ? gv[t] : (tbase + (c - 2) * 32 + t * 16 + r16);
            const unsigned short* kr = kb + (size_t)krow * D_;
#pragma unroll
            for (int kq = 0; kq < 4; ++kq) {
                FragU ubk;
                ubk.s.lo = *(const u16x8*)(kr + kq * 32 + hl * 8);
                ubk.s.hi = *(const u16x8*)(kr + kq * 32 + 16 + hl * 8);
                st[t] = __builtin_amdgcn_wmma_f32_16x16x32_bf16(
                    false, uq[kq].v, false, ubk.v, (short)0, st[t], false, false);
            }
        }

#pragma unroll
        for (int t = 0; t < 2; ++t) {
            float amv = 0.0f;
            int col = 0;
            if (!isg) {
                col = (c - 2) * 32 + t * 16 + r16;
                amv = am[(tbase + col) & (SEQ_ - 1)];
            }
#pragma unroll
            for (int i = 0; i < 8; ++i) {
                int row  = hl * 8 + i;
                float sv = st[t][i] * SCALE_;
                if (isg) {
                    st[t][i] = (gv[t] < tbase + row) ? sv : MINV_;
                } else {
                    int rel = col - row;
                    st[t][i] = (rel >= 1 && rel <= WIN_) ? (sv + amv) : MINV_;
                }
            }
        }

        float cm[8];
#pragma unroll
        for (int i = 0; i < 8; ++i) cm[i] = fmaxf(st[0][i], st[1][i]);
#pragma unroll
        for (int i = 0; i < 8; ++i)
#pragma unroll
            for (int off = 8; off >= 1; off >>= 1)
                cm[i] = fmaxf(cm[i], __shfl_xor(cm[i], off));

        float alpha[8];
#pragma unroll
        for (int i = 0; i < 8; ++i) {
            float mn = fmaxf(mrow[i], cm[i]);
            alpha[i] = __expf(mrow[i] - mn);
            mrow[i]  = mn;
            lrow[i] *= alpha[i];
        }
#pragma unroll
        for (int nt = 0; nt < 8; ++nt)
#pragma unroll
            for (int i = 0; i < 8; ++i) oacc[nt][i] *= alpha[i];

        float rs[8];
#pragma unroll
        for (int i = 0; i < 8; ++i) {
            float p0 = __expf(st[0][i] - mrow[i]);
            float p1 = __expf(st[1][i] - mrow[i]);
            rs[i] = p0 + p1;
            Pl[w][hl * 8 + i][r16]      = f32_to_bf16(p0);
            Pl[w][hl * 8 + i][16 + r16] = f32_to_bf16(p1);
        }
#pragma unroll
        for (int i = 0; i < 8; ++i) {
#pragma unroll
            for (int off = 8; off >= 1; off >>= 1)
                rs[i] += __shfl_xor(rs[i], off);
            lrow[i] += rs[i];
        }

        FragU up;
        up.s.lo = *(const u16x8*)&Pl[w][r16][hl * 8];
        up.s.hi = *(const u16x8*)&Pl[w][r16][16 + hl * 8];
#pragma unroll
        for (int nt = 0; nt < 8; ++nt) {
            int d = nt * 16 + r16;
            FragU ubv;
            if (isg) {
                ubv.s.lo = *(const u16x8*)&VgT[d][c * 32 + hl * 8];
                ubv.s.hi = *(const u16x8*)&VgT[d][c * 32 + 16 + hl * 8];
            } else {
                const unsigned short* vr =
                    vt + (size_t)d * PADS + (tbase + (c - 2) * 32);
                ubv.s.lo = *(const u16x8*)(vr + hl * 8);
                ubv.s.hi = *(const u16x8*)(vr + 16 + hl * 8);
            }
            oacc[nt] = __builtin_amdgcn_wmma_f32_16x16x32_bf16(
                false, up.v, false, ubv.v, (short)0, oacc[nt], false, false);
        }
    }

#pragma unroll
    for (int nt = 0; nt < 8; ++nt)
#pragma unroll
        for (int i = 0; i < 8; ++i) {
            int row  = hl * 8 + i;
            int qpos = WIN_ + tbase + row;
            int d    = nt * 16 + r16;
            out[(size_t)(b * SEQ_ + qpos) * HD_ + h * D_ + d] = oacc[nt][i] / lrow[i];
        }
}

// ---------------------------------------------------------------- launch
extern "C" void kernel_launch(void* const* d_in, const int* in_sizes, int n_in,
                              void* d_out, int out_size, void* d_ws, size_t ws_size,
                              hipStream_t stream) {
    (void)in_sizes; (void)n_in; (void)out_size; (void)ws_size;
    const float* hidden = (const float*)d_in[0];
    const float* amask  = (const float*)d_in[1];
    const int*   gidx   = (const int*)d_in[2];
    const float* Wqkv   = (const float*)d_in[3];
    const float* bqkv   = (const float*)d_in[4];
    const float* Wo     = (const float*)d_in[5];
    const float* bo     = (const float*)d_in[6];
    float* out = (float*)d_out;

    char* ws = (char*)d_ws;
    float*          qkv   = (float*)(ws);                          // 100,663,296 B
    float*          kmean = (float*)(ws + 100663296ULL);           //   2,097,152 B
    float*          vmean = (float*)(ws + 102760448ULL);           //   2,097,152 B
    float*          attnf = (float*)(ws + 104857600ULL);           //  33,554,432 B
    unsigned short* bfA   = (unsigned short*)(ws + 138412032ULL);  //  16,777,216 B
    unsigned short* bfB   = (unsigned short*)(ws + 155189248ULL);  //  25,165,824 B
    unsigned short* qbf   = (unsigned short*)(ws + 180355072ULL);  //  16,777,216 B
    unsigned short* kbf   = (unsigned short*)(ws + 197132288ULL);  //   1,064,960 B
    unsigned short* vbfT  = (unsigned short*)(ws + 198197248ULL);  //   1,064,960 B

    k_f32_to_bf16<<<2048, 256, 0, stream>>>(hidden, bfA, (long long)B_ * SEQ_ * HD_);
    k_f32_to_bf16_T<<<dim3(QKVLD / 32, HD_ / 32), dim3(32, 8), 0, stream>>>(
        Wqkv, bfB, HD_, QKVLD);
    dim3 g1(QKVLD / BN, (B_ * SEQ_) / BM);
    k_gemm_bf16<<<g1, 256, 0, stream>>>(bfA, bfB, bqkv, qkv, B_ * SEQ_, QKVLD, HD_);
    k_rope_mean<<<B_ * SEQ_, 128, 0, stream>>>(qkv, kmean, vmean);
    k_prep_q<<<B_ * H_ * SEQ_, 128, 0, stream>>>(qkv, qbf);
    k_prep_kv<<<B_ * PADS, 128, 0, stream>>>(kmean, vmean, kbf, vbfT);
    k_attn_prefix<<<B_ * H_ * (WIN_ / 8), 256, 0, stream>>>(qkv, kmean, vmean, amask, attnf);
    k_attn_steps_wmma<<<dim3(SQ_ / 16, 2, B_), 256, 0, stream>>>(
        qbf, kbf, vbfT, amask, gidx, attnf);
    k_f32_to_bf16<<<2048, 256, 0, stream>>>(attnf, bfA, (long long)B_ * SEQ_ * HD_);
    k_f32_to_bf16_T<<<dim3(HD_ / 32, HD_ / 32), dim3(32, 8), 0, stream>>>(
        Wo, bfB, HD_, HD_);
    dim3 g2(HD_ / BN, (B_ * SEQ_) / BM);
    k_gemm_bf16<<<g2, 256, 0, stream>>>(bfA, bfB, bo, out, B_ * SEQ_, HD_, HD_);
}
